// MultiHeadAttention_87754771792294
// MI455X (gfx1250) — matrix operand order repaired and checked by reading, not compiled, not run on hardware
//
#include <hip/hip_runtime.h>

typedef __bf16 bf16;
typedef __bf16 v16bf __attribute__((ext_vector_type(16)));
typedef float  v8f   __attribute__((ext_vector_type(8)));

#define E_DIM 768
#define HEADS 8
#define HDIM  96
#define NSEQ  1024
#define BATCH 16

// ---------------------------------------------------------------------------
// CDNA5 async memory->LDS copy (16 bytes per lane), tracked by ASYNCcnt.
// ---------------------------------------------------------------------------
__device__ __forceinline__ void async_copy16(unsigned int lds_off, const void* gptr) {
    asm volatile("global_load_async_to_lds_b128 %0, %1, off"
                 :: "v"(lds_off), "v"(gptr) : "memory");
}

template <int N>
__device__ __forceinline__ void wait_async() {
    asm volatile("s_wait_asynccnt %0" :: "n"(N) : "memory");
}

__device__ __forceinline__ unsigned int lds_off32(const void* p) {
    // generic pointer to LDS: low 32 bits are the in-allocation byte offset
    return (unsigned int)(size_t)p;
}

// ---------------------------------------------------------------------------
// Kernel 1a: f32 -> bf16 elementwise convert
// ---------------------------------------------------------------------------
__global__ __launch_bounds__(256) void cvt_f32_bf16(const float* __restrict__ src,
                                                    bf16* __restrict__ dst, int n) {
    int i = blockIdx.x * 256 + threadIdx.x;
    if (i < n) dst[i] = (bf16)src[i];
}

// ---------------------------------------------------------------------------
// Kernel 1b: transpose 768x768 f32 W (K x N) into bf16 W^T (N x K)
// ---------------------------------------------------------------------------
__global__ __launch_bounds__(256) void transpose_w(const float* __restrict__ W,
                                                   bf16* __restrict__ Wt) {
    int k = blockIdx.y;
    int n = blockIdx.x * 256 + threadIdx.x;
    if (n < E_DIM) Wt[(size_t)n * E_DIM + k] = (bf16)W[(size_t)k * E_DIM + n];
}

// ---------------------------------------------------------------------------
// Kernel 2: C[M,N] = A[M,K] @ W[K,N] + bias, W given as W^T (N x K) bf16.
// Block: 256 thr = 8 waves, tile 128(M) x 128(N), K-step 64, double-buffered
// async-to-LDS staging (64 KB LDS). Wave w = (wM = w&3, wN = w>>2) owns
// 32x64 = 2x4 WMMA tiles -> 16 WMMAs per wave per K-step.
// Output bf16 (Cb) or f32 (Cf).
// ---------------------------------------------------------------------------
#define TM 128
#define TN 128
#define TK 64

__global__ __launch_bounds__(256) void gemm_bf16(const bf16* __restrict__ A,
                                                 const bf16* __restrict__ Wt,
                                                 const float* __restrict__ bias,
                                                 bf16* __restrict__ Cb,
                                                 float* __restrict__ Cf,
                                                 int M, int K, int N) {
    __shared__ bf16 As[2][TM][TK];   // 2 x 16 KB
    __shared__ bf16 Bs[2][TN][TK];   // 2 x 16 KB

    const int tid  = threadIdx.x;
    const int lane = tid & 31;
    const int wave = tid >> 5;
    const int wM   = wave & 3;       // 4 waves along M (32 rows each)
    const int wN   = wave >> 2;      // 2 waves along N (64 cols each)
    const int col  = lane & 15;
    const int kg   = lane >> 4;

    const int m0 = blockIdx.y * TM;
    const int n0 = blockIdx.x * TN;

    // staging: 256 threads, each copies 64 B of A and 64 B of B per step
    const int sr = tid >> 1;             // row 0..127
    const int sc = (tid & 1) * 32;       // bf16 col 0 or 32

    const v8f zero8 = {0.f,0.f,0.f,0.f,0.f,0.f,0.f,0.f};
    v8f acc[2][4];
    #pragma unroll
    for (int i = 0; i < 2; ++i)
        #pragma unroll
        for (int j = 0; j < 4; ++j) acc[i][j] = zero8;

    auto stage = [&](int buf, int k0) {
        const bf16* sa = A  + (size_t)(m0 + sr) * K + k0 + sc;
        const bf16* sb = Wt + (size_t)(n0 + sr) * K + k0 + sc;
        unsigned int la = lds_off32(&As[buf][sr][sc]);
        unsigned int lb = lds_off32(&Bs[buf][sr][sc]);
        #pragma unroll
        for (int c = 0; c < 4; ++c) {
            async_copy16(la + 16 * c, sa + 8 * c);
            async_copy16(lb + 16 * c, sb + 8 * c);
        }
    };

    auto compute = [&](int buf) {
        #pragma unroll
        for (int kk = 0; kk < TK; kk += 32) {
            v16bf af[2], bfr[4];
            #pragma unroll
            for (int i = 0; i < 2; ++i) {
                const bf16* ap = &As[buf][wM * 32 + i * 16 + col][kk + kg * 16];
                #pragma unroll
                for (int e = 0; e < 16; ++e) af[i][e] = ap[e];
            }
            #pragma unroll
            for (int j = 0; j < 4; ++j) {
                const bf16* bp = &Bs[buf][wN * 64 + j * 16 + col][kk + kg * 16];
                #pragma unroll
                for (int e = 0; e < 16; ++e) bfr[j][e] = bp[e];
            }
            #pragma unroll
            for (int i = 0; i < 2; ++i)
                #pragma unroll
                for (int j = 0; j < 4; ++j)
                    acc[i][j] = __builtin_amdgcn_wmma_f32_16x16x32_bf16(
                        false, af[i], false, bfr[j], (short)0, acc[i][j], false, false);
        }
    };

    // peeled double-buffered pipeline: branch-free steady state
    stage(0, 0);
    int buf = 0;
    for (int k0 = 0; k0 + TK < K; k0 += TK) {
        stage(buf ^ 1, k0 + TK);
        wait_async<8>();
        __syncthreads();
        compute(buf);
        __syncthreads();
        buf ^= 1;
    }
    wait_async<0>();
    __syncthreads();
    compute(buf);

    // ---- epilogue: bias + store (C layout: vgpr v -> row v+8*kg, lane&15 -> col) ----
    const int rowOff = kg * 8;
    #pragma unroll
    for (int j = 0; j < 4; ++j) {
        const int n = n0 + wN * 64 + j * 16 + col;
        const float bv = bias[n];
        #pragma unroll
        for (int i = 0; i < 2; ++i) {
            #pragma unroll
            for (int v = 0; v < 8; ++v) {
                const int m = m0 + wM * 32 + i * 16 + rowOff + v;
                const float val = acc[i][j][v] + bv;
                if (Cf) Cf[(size_t)m * N + n] = val;
                else    Cb[(size_t)m * N + n] = (bf16)val;
            }
        }
    }
}

// ---------------------------------------------------------------------------
// Kernel 3: flash attention per (b,h), double-buffered async K/V panels.
// grid = (NSEQ/64, BATCH*HEADS), block = 128 thr = 4 waves; wave owns 16 rows.
// Reference semantics: attn = softmax(Q K^T) / sqrt(E); out = attn @ V.
// ---------------------------------------------------------------------------
__global__ __launch_bounds__(128) void attn_flash(const bf16* __restrict__ Q,
                                                  const bf16* __restrict__ Kmat,
                                                  const bf16* __restrict__ V,
                                                  bf16* __restrict__ O) {
    __shared__ bf16 Ks[2][32][HDIM];    // 2 x 6 KB
    __shared__ bf16 Vs[2][32][HDIM];    // 2 x 6 KB
    __shared__ bf16 Pbuf[4][16][32];    // 4 KB (per-wave P relayout C->A)

    const int bh   = blockIdx.y;
    const int b    = bh >> 3;
    const int h    = bh & 7;
    const int tid  = threadIdx.x;
    const int lane = tid & 31;
    const int wave = tid >> 5;
    const int col  = lane & 15;
    const int kg   = lane >> 4;

    const int qrow0   = blockIdx.x * 64 + wave * 16;
    const size_t base = (size_t)b * NSEQ * E_DIM + (size_t)h * HDIM;

    // stage one 32-key K/V panel: 32 rows x 192 B = 384 16B-chunks each,
    // 128 threads x 3 chunks, all 16B aligned.
    auto stageKV = [&](int buf, int kb) {
        #pragma unroll
        for (int j = 0; j < 3; ++j) {
            const int chunk = tid + j * 128;      // 0..383
            const int row   = chunk / 12;         // 12 chunks per row
            const int cb    = (chunk % 12) * 16;  // byte offset within row
            const size_t g  = base + (size_t)(kb * 32 + row) * E_DIM;
            async_copy16(lds_off32(&Ks[buf][row][0]) + cb, (const char*)(Kmat + g) + cb);
            async_copy16(lds_off32(&Vs[buf][row][0]) + cb, (const char*)(V    + g) + cb);
        }
    };

    stageKV(0, 0);

    // ---- preload Q A-fragments: 16 rows x 96 = 3 chunks of K=32 ----
    v16bf qf[3];
    {
        const int qr = qrow0 + col;
        #pragma unroll
        for (int c = 0; c < 3; ++c) {
            const bf16* qp = Q + base + (size_t)qr * E_DIM + c * 32 + kg * 16;
            #pragma unroll
            for (int e = 0; e < 16; ++e) qf[c][e] = qp[e];
        }
    }

    const v8f zero8 = {0.f,0.f,0.f,0.f,0.f,0.f,0.f,0.f};
    v8f acc[6] = {zero8, zero8, zero8, zero8, zero8, zero8};
    float mrow[8], lrow[8];
    #pragma unroll
    for (int v = 0; v < 8; ++v) { mrow[v] = -1e30f; lrow[v] = 0.f; }

    int buf = 0;
    for (int kb = 0; kb < NSEQ / 32; ++kb) {
        const bool more = (kb + 1) < NSEQ / 32;
        if (more) { stageKV(buf ^ 1, kb + 1); wait_async<6>(); }
        else      { wait_async<0>(); }
        __syncthreads();

        // ---- S = Q K^T : two 16-key tiles, accumulate over D = 3x32 ----
        v8f s[2];
        #pragma unroll
        for (int t = 0; t < 2; ++t) {
            s[t] = zero8;
            #pragma unroll
            for (int c = 0; c < 3; ++c) {
                v16bf kf;
                const bf16* kp = &Ks[buf][t * 16 + col][c * 32 + kg * 16];
                #pragma unroll
                for (int e = 0; e < 16; ++e) kf[e] = kp[e];
                s[t] = __builtin_amdgcn_wmma_f32_16x16x32_bf16(
                    false, qf[c], false, kf, (short)0, s[t], false, false);
            }
        }

        // ---- online softmax stats (row = 16 lanes of one half-wave) ----
        float corr[8];
        #pragma unroll
        for (int v = 0; v < 8; ++v) {
            float r = fmaxf(s[0][v], s[1][v]);
            #pragma unroll
            for (int off = 8; off >= 1; off >>= 1)
                r = fmaxf(r, __shfl_xor(r, off, 32));
            const float mn = fmaxf(mrow[v], r);
            corr[v] = __expf(mrow[v] - mn);
            mrow[v] = mn;
            lrow[v] *= corr[v];
        }
        #pragma unroll
        for (int d = 0; d < 6; ++d)
            #pragma unroll
            for (int v = 0; v < 8; ++v) acc[d][v] *= corr[v];

        #pragma unroll
        for (int v = 0; v < 8; ++v) {
            const float p0 = __expf(s[0][v] - mrow[v]);
            const float p1 = __expf(s[1][v] - mrow[v]);
            s[0][v] = p0; s[1][v] = p1;
            float r = p0 + p1;
            #pragma unroll
            for (int off = 8; off >= 1; off >>= 1)
                r += __shfl_xor(r, off, 32);
            lrow[v] += r;
        }

        // ---- relayout P (C layout -> A layout) through LDS ----
        #pragma unroll
        for (int t = 0; t < 2; ++t)
            #pragma unroll
            for (int v = 0; v < 8; ++v)
                Pbuf[wave][v + 8 * kg][t * 16 + col] = (bf16)s[t][v];
        __syncthreads();

        v16bf pf;
        {
            const bf16* pp = &Pbuf[wave][col][kg * 16];
            #pragma unroll
            for (int e = 0; e < 16; ++e) pf[e] = pp[e];
        }

        // ---- acc += P @ V : 6 d-tiles ----
        #pragma unroll
        for (int dt = 0; dt < 6; ++dt) {
            v16bf vf;
            #pragma unroll
            for (int e = 0; e < 16; ++e) vf[e] = Vs[buf][kg * 16 + e][dt * 16 + col];
            acc[dt] = __builtin_amdgcn_wmma_f32_16x16x32_bf16(
                false, pf, false, vf, (short)0, acc[dt], false, false);
        }
        __syncthreads();   // all waves done with buf before it is restaged
        buf ^= 1;
    }

    // ---- epilogue: out = acc / l / sqrt(E) (non-standard scaling folded) ----
    const float inv_sqrtE = 0.036084391824351615f;   // 1/sqrt(768)
    const int rowOff = kg * 8;
    #pragma unroll
    for (int v = 0; v < 8; ++v) {
        const float sc = inv_sqrtE / lrow[v];
        const int r = qrow0 + rowOff + v;
        #pragma unroll
        for (int dt = 0; dt < 6; ++dt)
            O[base + (size_t)r * E_DIM + dt * 16 + col] = (bf16)(acc[dt][v] * sc);
    }
}

// ---------------------------------------------------------------------------
// Launch
// ---------------------------------------------------------------------------
extern "C" void kernel_launch(void* const* d_in, const int* in_sizes, int n_in,
                              void* d_out, int out_size, void* d_ws, size_t ws_size,
                              hipStream_t stream) {
    const float* x  = (const float*)d_in[0];
    const float* Wq = (const float*)d_in[1];
    const float* bq = (const float*)d_in[2];
    const float* Wk = (const float*)d_in[3];
    const float* bk = (const float*)d_in[4];
    const float* Wv = (const float*)d_in[5];
    const float* bv = (const float*)d_in[6];
    const float* Wo = (const float*)d_in[7];
    const float* bo = (const float*)d_in[8];
    float* out = (float*)d_out;

    const size_t M  = (size_t)BATCH * NSEQ;      // 16384
    const size_t XE = M * E_DIM;
    const size_t WE = (size_t)E_DIM * E_DIM;

    char* ws = (char*)d_ws;
    size_t off = 0;
    bf16* xb  = (bf16*)(ws + off); off += XE * 2;
    bf16* wqT = (bf16*)(ws + off); off += WE * 2;
    bf16* wkT = (bf16*)(ws + off); off += WE * 2;
    bf16* wvT = (bf16*)(ws + off); off += WE * 2;
    bf16* woT = (bf16*)(ws + off); off += WE * 2;
    bf16* Qb  = (bf16*)(ws + off); off += XE * 2;
    bf16* Kb  = (bf16*)(ws + off); off += XE * 2;
    bf16* Vb  = (bf16*)(ws + off); off += XE * 2;
    bf16* AO  = (bf16*)(ws + off); off += XE * 2;   // ~131 MB total

    cvt_f32_bf16<<<(int)((XE + 255) / 256), 256, 0, stream>>>(x, xb, (int)XE);
    dim3 tg(E_DIM / 256, E_DIM);
    transpose_w<<<tg, 256, 0, stream>>>(Wq, wqT);
    transpose_w<<<tg, 256, 0, stream>>>(Wk, wkT);
    transpose_w<<<tg, 256, 0, stream>>>(Wv, wvT);
    transpose_w<<<tg, 256, 0, stream>>>(Wo, woT);

    dim3 ggrid(E_DIM / TN, (int)(M / TM));       // (6, 128)
    gemm_bf16<<<ggrid, 256, 0, stream>>>(xb, wqT, bq, Qb, nullptr, (int)M, E_DIM, E_DIM);
    gemm_bf16<<<ggrid, 256, 0, stream>>>(xb, wkT, bk, Kb, nullptr, (int)M, E_DIM, E_DIM);
    gemm_bf16<<<ggrid, 256, 0, stream>>>(xb, wvT, bv, Vb, nullptr, (int)M, E_DIM, E_DIM);

    dim3 agrid(NSEQ / 64, BATCH * HEADS);        // (16, 128)
    attn_flash<<<agrid, 128, 0, stream>>>(Qb, Kb, Vb, AO);

    gemm_bf16<<<ggrid, 256, 0, stream>>>(AO, woT, bo, nullptr, out, (int)M, E_DIM, E_DIM);
}
